// MLA_66099546685732
// MI455X (gfx1250) — compile-verified, matrix operand order checked by reading, not run on hardware
//
#include <hip/hip_runtime.h>
#include <hip/hip_bf16.h>
#include <stdint.h>

// ---------------- problem constants (match reference) ----------------
#define B_    2
#define S_    2048
#define DIM_  2048
#define H_    16
#define NOPE_ 128
#define ROPE_ 64
#define VHD_  128
#define KLR_  512
#define QKHD_ (NOPE_ + ROPE_)   // 192
#define KCAT_ (KLR_ + ROPE_)    // 576
#define SCALE_ 0.07216878364870322f   // 192^-0.5
#define EPS_   1.1920929e-07f

// ---------------- types ----------------
typedef __attribute__((ext_vector_type(16))) __bf16        v16bf;
typedef __attribute__((ext_vector_type(8)))  float         v8f;
typedef __attribute__((ext_vector_type(4)))  unsigned int  u32x4;

union Frag { u32x4 q[2]; v16bf v; };

__device__ __forceinline__ v8f wmma_bf16(const v16bf& a, const v16bf& b, const v8f& c) {
  // v_wmma_f32_16x16x32_bf16  (8-arg form: neg_a, A, neg_b, B, c_mod, C, reuse_a, reuse_b)
  return __builtin_amdgcn_wmma_f32_16x16x32_bf16(false, a, false, b, (short)0, c, false, false);
}

__device__ __forceinline__ unsigned short f2bf(float x) {
  unsigned int u = __float_as_uint(x);
  u += 0x7fffu + ((u >> 16) & 1u);            // round-to-nearest-even
  return (unsigned short)(u >> 16);
}

// =====================================================================
// Generic 128x128 tiled GEMM:  C[M][N] = A_f32[M][K] * B_bf16[N][K]^T
//   8 waves / block; each wave computes a 64x32 sub-tile (4x2 WMMA tiles).
//   MODE 0: write f32 C (ldc, batch offset cBatch per blockIdx.z)
//   MODE 1: q-projection store: row m = b*S+s  ->  qcat[((b*H+h)*S+s)*576+n]
//           as bf16, scaled by `scale` (h = blockIdx.z)
// =====================================================================
template <int MODE>
__global__ __launch_bounds__(256)
void gemm128_kernel(const float* __restrict__ A, int lda, long aBatch,
                    const unsigned short* __restrict__ Bm, int ldb, long bBatch,
                    float* __restrict__ C, int ldc, long cBatch,
                    unsigned short* __restrict__ Cb, float scale,
                    int M, int N, int K)
{
  __shared__ unsigned short As[128][40];   // 128 x 32 bf16 tile (+pad)
  __shared__ unsigned short Bs[128][40];

  const int z = blockIdx.z;
  A  += (long)z * aBatch;
  Bm += (long)z * bBatch;
  if (MODE == 0) C += (long)z * cBatch;

  const int m0   = blockIdx.y * 128;
  const int n0   = blockIdx.x * 128;
  const int tid  = threadIdx.x;
  const int lane = tid & 31;
  const int w    = tid >> 5;
  const int wm   = w >> 2;                 // 0..1  (64 rows each)
  const int wn   = w & 3;                  // 0..3  (32 cols each)
  const int l15  = lane & 15;
  const int ko1  = (lane >> 4) << 3;       // 0 or 8 (A/B fragment K pattern)

  v8f vzero = {0.f,0.f,0.f,0.f,0.f,0.f,0.f,0.f};
  v8f acc[4][2];
  #pragma unroll
  for (int i = 0; i < 4; ++i)
    #pragma unroll
    for (int j = 0; j < 2; ++j) acc[i][j] = vzero;

  const int srow = tid >> 1;               // 0..127
  const int scol = (tid & 1) * 16;         // 0 or 16

  for (int k0 = 0; k0 < K; k0 += 32) {
    // ---- stage A (convert f32 -> bf16) and B (already bf16) into LDS ----
    {
      const int gr = m0 + srow;
      #pragma unroll
      for (int c = 0; c < 16; ++c) {
        const int gc = k0 + scol + c;
        float v = (gr < M && gc < K) ? A[(long)gr * lda + gc] : 0.0f;
        As[srow][scol + c] = f2bf(v);
      }
      const int gn = n0 + srow;
      #pragma unroll
      for (int c = 0; c < 16; ++c) {
        const int gc = k0 + scol + c;
        Bs[srow][scol + c] =
            (gn < N && gc < K) ? Bm[(long)gn * ldb + gc] : (unsigned short)0;
      }
    }
    __syncthreads();

    Frag a[4], b[2];
    #pragma unroll
    for (int mt = 0; mt < 4; ++mt) {
      const u32x4* p = (const u32x4*)&As[wm * 64 + mt * 16 + l15][0];
      a[mt].q[0] = p[ko1 >> 3];
      a[mt].q[1] = p[(ko1 + 16) >> 3];
    }
    #pragma unroll
    for (int nt = 0; nt < 2; ++nt) {
      const u32x4* p = (const u32x4*)&Bs[wn * 32 + nt * 16 + l15][0];
      b[nt].q[0] = p[ko1 >> 3];
      b[nt].q[1] = p[(ko1 + 16) >> 3];
    }
    #pragma unroll
    for (int mt = 0; mt < 4; ++mt)
      #pragma unroll
      for (int nt = 0; nt < 2; ++nt)
        acc[mt][nt] = wmma_bf16(a[mt].v, b[nt].v, acc[mt][nt]);
    __syncthreads();
  }

  // ---- store (C/D layout: VGPR r -> row r (lanes 0-15) / r+8 (lanes 16-31)) ----
  const int rbase = (lane >= 16) ? 8 : 0;
  #pragma unroll
  for (int mt = 0; mt < 4; ++mt) {
    #pragma unroll
    for (int nt = 0; nt < 2; ++nt) {
      #pragma unroll
      for (int r = 0; r < 8; ++r) {
        const int row = m0 + wm * 64 + mt * 16 + rbase + r;
        const int col = n0 + wn * 32 + nt * 16 + l15;
        if (row < M && col < N) {
          if (MODE == 0) {
            C[(long)row * ldc + col] = acc[mt][nt][r];
          } else {
            const int bb = row / S_;
            const int ss = row - bb * S_;
            const long orow = (long)(bb * H_ + z) * S_ + ss;
            Cb[orow * KCAT_ + col] = f2bf(acc[mt][nt][r] * scale);
          }
        }
      }
    }
  }
}

// =====================================================================
// RMSNorm(kv_c)*w + RoPE(k_pe)  ->  kcat[b,t,576] (bf16) and kv_c^T (bf16)
// one block per (b, t) row
// =====================================================================
__global__ __launch_bounds__(256)
void prep_kv_kernel(const float* __restrict__ kvraw,
                    const float* __restrict__ kvw,
                    const float* __restrict__ fcos,
                    const float* __restrict__ fsin,
                    unsigned short* __restrict__ kcat,
                    unsigned short* __restrict__ kvcT)
{
  __shared__ float red[256];
  const int row = blockIdx.x;              // b*S + s
  const int b = row / S_, s = row - b * S_;
  const int tid = threadIdx.x;
  const float* src = kvraw + (long)row * KCAT_;

  float ss = 0.f;
  for (int i = tid; i < KLR_; i += 256) { float v = src[i]; ss += v * v; }
  red[tid] = ss;
  __syncthreads();
  for (int off = 128; off > 0; off >>= 1) {
    if (tid < off) red[tid] += red[tid + off];
    __syncthreads();
  }
  const float rms = rsqrtf(red[0] / (float)KLR_ + EPS_);

  for (int i = tid; i < KLR_; i += 256) {
    const unsigned short hv = f2bf(src[i] * rms * kvw[i]);
    kcat[(long)row * KCAT_ + i] = hv;
    kvcT[(long)(b * KLR_ + i) * S_ + s] = hv;
  }
  if (tid < 32) {
    const int p = tid;
    const float xr = src[KLR_ + 2 * p], xi = src[KLR_ + 2 * p + 1];
    const float c = fcos[s * 32 + p], sn = fsin[s * 32 + p];
    kcat[(long)row * KCAT_ + KLR_ + 2 * p]     = f2bf(xr * c - xi * sn);
    kcat[(long)row * KCAT_ + KLR_ + 2 * p + 1] = f2bf(xr * sn + xi * c);
  }
}

// RoPE(q_pe) * SCALE  ->  qcat[b,h,s,512:576]
__global__ void prep_q_kernel(const float* __restrict__ q,
                              const float* __restrict__ fcos,
                              const float* __restrict__ fsin,
                              unsigned short* __restrict__ qcat)
{
  const long i = (long)blockIdx.x * blockDim.x + threadIdx.x;
  const long total = (long)B_ * S_ * H_ * 32;
  if (i >= total) return;
  const int p = (int)(i & 31);
  const int h = (int)((i >> 5) & (H_ - 1));
  const long t = i >> 9;
  const int s = (int)(t % S_);
  const int b = (int)(t / S_);
  const float* qp = q + (long)(b * S_ + s) * (H_ * QKHD_) + h * QKHD_ + NOPE_;
  const float xr = qp[2 * p], xi = qp[2 * p + 1];
  const float c = fcos[s * 32 + p], sn = fsin[s * 32 + p];
  unsigned short* dst = qcat + ((long)(b * H_ + h) * S_ + s) * (long)KCAT_ + KLR_;
  dst[2 * p]     = f2bf((xr * c - xi * sn) * SCALE_);
  dst[2 * p + 1] = f2bf((xr * sn + xi * c) * SCALE_);
}

// ---------------- weight conversions ----------------
__global__ void cvt_bf16_kernel(const float* __restrict__ src,
                                unsigned short* __restrict__ dst, long n) {
  const long i = (long)blockIdx.x * blockDim.x + threadIdx.x;
  if (i < n) dst[i] = f2bf(src[i]);
}
// wb1t[h][c=0..511][d=0..127] = wkv_b[h*256 + d][c]   (transposed for [N][K] GEMM)
__global__ void cvt_wb1t_kernel(const float* __restrict__ wkvb,
                                unsigned short* __restrict__ dst) {
  const long i = (long)blockIdx.x * blockDim.x + threadIdx.x;
  const long n = (long)H_ * KLR_ * NOPE_;
  if (i >= n) return;
  const int h = (int)(i / (KLR_ * NOPE_));
  const int r = (int)(i % (KLR_ * NOPE_));
  const int c = r / NOPE_;
  const int d = r % NOPE_;
  dst[i] = f2bf(wkvb[(long)(h * (NOPE_ + VHD_) + d) * KLR_ + c]);
}
// wb2[h][d=0..127][c=0..511] = wkv_b[h*256 + 128 + d][c]  (already [N][K])
__global__ void cvt_wb2_kernel(const float* __restrict__ wkvb,
                               unsigned short* __restrict__ dst) {
  const long i = (long)blockIdx.x * blockDim.x + threadIdx.x;
  const long n = (long)H_ * VHD_ * KLR_;
  if (i >= n) return;
  const int h = (int)(i / (VHD_ * KLR_));
  const int r = (int)(i % (VHD_ * KLR_));
  const int d = r / KLR_;
  const int c = r % KLR_;
  dst[i] = f2bf(wkvb[(long)(h * (NOPE_ + VHD_) + NOPE_ + d) * KLR_ + c]);
}

// =====================================================================
// Flash attention in latent space.
//   grid (S/64, H, B), 256 threads = 8 waves.
//   Wave pair (qt) owns 16 query rows; the two waves (ch=0/1) split the
//   18 score k-steps (K=576) and each owns 256 of the 512 output columns.
//   K fragments from kcat[b,t,576], V fragments from kv_c^T[b,512,S]
//   stream straight from global (L2-resident on the 192MB L2).
//   PV fragments are loaded in batches of 4 to expose MLP; the next key
//   tile is prefetched (global_prefetch_b8) while the current one computes.
// =====================================================================
__global__ __launch_bounds__(256)
void attn_kernel(const unsigned short* __restrict__ qcat,
                 const unsigned short* __restrict__ kcat,
                 const unsigned short* __restrict__ kvcT,
                 float* __restrict__ o)
{
  __shared__ float          sm_sc[4][2][16][32];   // partial scores
  __shared__ unsigned short sm_p[4][16][32];       // P tile (bf16)
  __shared__ float          sm_f[4][16];           // per-row rescale factor
  __shared__ float          sm_l[4][16];           // per-row running sum

  const int b = blockIdx.z, h = blockIdx.y;
  const int qbase = blockIdx.x * 64;
  const int tid  = threadIdx.x;
  const int lane = tid & 31;
  const int w    = tid >> 5;
  const int qt   = w >> 1;                 // 0..3 : query 16-row tile
  const int ch   = w & 1;                  // 0/1  : k-split + column half
  const int l15  = lane & 15;
  const int koff = (lane >= 16) ? 8 : 0;
  const int rbase = (lane >= 16) ? 8 : 0;
  const int qr0 = qbase + qt * 16;

  const unsigned short* qrowp =
      qcat + ((long)(b * H_ + h) * S_ + qr0 + l15) * (long)KCAT_;

  v8f vzero = {0.f,0.f,0.f,0.f,0.f,0.f,0.f,0.f};
  v8f acc[16];
  #pragma unroll
  for (int i = 0; i < 16; ++i) acc[i] = vzero;

  float m_row = -3.0e38f, l_row = 0.0f;    // valid in ch==0, lane<16

  const int nkt = (qbase + 64) / 32;       // causal: key tiles up to diagonal
  for (int kt = 0; kt < nkt; ++kt) {
    const int tbase = kt * 32;

    // ---- prefetch next key tile (K rows + V columns) into cache ----
    if (kt + 1 < nkt) {
      const int ntb = tbase + 32;
      __builtin_prefetch(kcat + (long)(b * S_ + ntb + lane) * KCAT_, 0, 0);
      __builtin_prefetch(
          kvcT + (long)(b * KLR_ + ch * 256 + lane * 8) * S_ + ntb, 0, 0);
    }

    // ---- partial scores: 9 of 18 k-steps per wave, two 16x16 key tiles ----
    v8f s0 = vzero, s1 = vzero;
    #pragma unroll
    for (int ks = 0; ks < 9; ++ks) {
      const int kb = (ch * 9 + ks) * 32 + koff;
      Frag a;
      a.q[0] = *(const u32x4*)(qrowp + kb);
      a.q[1] = *(const u32x4*)(qrowp + kb + 16);
      const unsigned short* kp0 = kcat + (long)(b * S_ + tbase + l15)      * KCAT_ + kb;
      const unsigned short* kp1 = kcat + (long)(b * S_ + tbase + 16 + l15) * KCAT_ + kb;
      Frag b0, b1;
      b0.q[0] = *(const u32x4*)kp0;  b0.q[1] = *(const u32x4*)(kp0 + 16);
      b1.q[0] = *(const u32x4*)kp1;  b1.q[1] = *(const u32x4*)(kp1 + 16);
      s0 = wmma_bf16(a.v, b0.v, s0);
      s1 = wmma_bf16(a.v, b1.v, s1);
    }
    #pragma unroll
    for (int r = 0; r < 8; ++r) {
      sm_sc[qt][ch][rbase + r][l15]      = s0[r];
      sm_sc[qt][ch][rbase + r][16 + l15] = s1[r];
    }
    __syncthreads();

    // ---- online softmax: one lane per query row ----
    if (ch == 0 && lane < 16) {
      const int ri = lane;
      const int qrow = qr0 + ri;
      float mx = m_row;
      for (int j = 0; j < 32; ++j) {
        float v = sm_sc[qt][0][ri][j] + sm_sc[qt][1][ri][j];
        if (tbase + j > qrow) v = -1.0e9f;           // causal mask
        mx = fmaxf(mx, v);
      }
      const float f = __expf(m_row - mx);
      float sum = 0.0f;
      for (int j = 0; j < 32; ++j) {
        float v = sm_sc[qt][0][ri][j] + sm_sc[qt][1][ri][j];
        if (tbase + j > qrow) v = -1.0e9f;
        const float p = __expf(v - mx);
        sum += p;
        sm_p[qt][ri][j] = f2bf(p);
      }
      l_row = l_row * f + sum;
      m_row = mx;
      sm_f[qt][ri] = f;
      sm_l[qt][ri] = l_row;
    }
    __syncthreads();

    // ---- rescale accumulators, then O += P x V ----
    float fr[8];
    #pragma unroll
    for (int r = 0; r < 8; ++r) fr[r] = sm_f[qt][rbase + r];
    #pragma unroll
    for (int t = 0; t < 16; ++t)
      #pragma unroll
      for (int r = 0; r < 8; ++r) acc[t][r] *= fr[r];

    Frag pa;
    const unsigned short* pp = &sm_p[qt][l15][0];
    pa.q[0] = *(const u32x4*)(pp + koff);
    pa.q[1] = *(const u32x4*)(pp + koff + 16);

    // 16 independent V fragments: load in batches of 4, then 4 WMMAs,
    // so the compiler can clause 8 global_load_b128 and hide latency.
    #pragma unroll
    for (int nb = 0; nb < 4; ++nb) {
      Frag vb[4];
      #pragma unroll
      for (int j = 0; j < 4; ++j) {
        const int ccol = ch * 256 + (nb * 4 + j) * 16 + l15;
        const unsigned short* vp =
            kvcT + (long)(b * KLR_ + ccol) * S_ + tbase + koff;
        vb[j].q[0] = *(const u32x4*)vp;
        vb[j].q[1] = *(const u32x4*)(vp + 16);
      }
      #pragma unroll
      for (int j = 0; j < 4; ++j)
        acc[nb * 4 + j] = wmma_bf16(pa.v, vb[j].v, acc[nb * 4 + j]);
    }
    __syncthreads();
  }

  // ---- normalize and store o[b,s,h,512] (f32) ----
  float lr[8];
  #pragma unroll
  for (int r = 0; r < 8; ++r) lr[r] = 1.0f / sm_l[qt][rbase + r];
  #pragma unroll
  for (int nt = 0; nt < 16; ++nt) {
    const int ccol = ch * 256 + nt * 16 + l15;
    #pragma unroll
    for (int r = 0; r < 8; ++r) {
      const int srow = qr0 + rbase + r;
      o[(long)(b * S_ + srow) * (H_ * KLR_) + h * KLR_ + ccol] = acc[nt][r] * lr[r];
    }
  }
}

// =====================================================================
// Host-side orchestration
// =====================================================================
extern "C" void kernel_launch(void* const* d_in, const int* in_sizes, int n_in,
                              void* d_out, int out_size, void* d_ws, size_t ws_size,
                              hipStream_t stream)
{
  const float* x     = (const float*)d_in[0];
  const float* wq    = (const float*)d_in[1];
  const float* wkv_a = (const float*)d_in[2];
  const float* kvw   = (const float*)d_in[3];
  const float* wkv_b = (const float*)d_in[4];
  const float* wo    = (const float*)d_in[5];
  const float* fcos  = (const float*)d_in[6];
  const float* fsin  = (const float*)d_in[7];
  // d_in[8] = mask: applied analytically (causal, 0 / -1e9), not read.
  float* out = (float*)d_out;
  (void)in_sizes; (void)n_in; (void)out_size; (void)ws_size;

  char* wp = (char*)d_ws;
  auto take = [&](size_t bytes) -> char* {
    char* p = wp;
    wp += (bytes + 255) & ~(size_t)255;
    return p;
  };

  unsigned short* wq_bf   = (unsigned short*)take((size_t)H_ * QKHD_ * DIM_ * 2);
  unsigned short* wkva_bf = (unsigned short*)take((size_t)KCAT_ * DIM_ * 2);
  unsigned short* wo_bf   = (unsigned short*)take((size_t)DIM_ * H_ * VHD_ * 2);
  unsigned short* wb1t    = (unsigned short*)take((size_t)H_ * KLR_ * NOPE_ * 2);
  unsigned short* wb2     = (unsigned short*)take((size_t)H_ * VHD_ * KLR_ * 2);
  float*          qf      = (float*)take((size_t)B_ * S_ * H_ * QKHD_ * 4);
  float*          kvraw   = (float*)take((size_t)B_ * S_ * KCAT_ * 4);
  unsigned short* qcat    = (unsigned short*)take((size_t)B_ * H_ * S_ * KCAT_ * 2);
  unsigned short* kcat    = (unsigned short*)take((size_t)B_ * S_ * KCAT_ * 2);
  unsigned short* kvcT    = (unsigned short*)take((size_t)B_ * KLR_ * S_ * 2);
  float*          oat     = (float*)take((size_t)B_ * S_ * H_ * KLR_ * 4);
  float*          o2      = (float*)take((size_t)B_ * S_ * H_ * VHD_ * 4);

  const int M = B_ * S_;   // 4096 token rows

  // ---- weight conversions to bf16 ----
  {
    const long n1 = (long)H_ * QKHD_ * DIM_;
    cvt_bf16_kernel<<<dim3((unsigned)((n1 + 255) / 256)), 256, 0, stream>>>(wq, wq_bf, n1);
    const long n2 = (long)KCAT_ * DIM_;
    cvt_bf16_kernel<<<dim3((unsigned)((n2 + 255) / 256)), 256, 0, stream>>>(wkv_a, wkva_bf, n2);
    const long n3 = (long)DIM_ * H_ * VHD_;
    cvt_bf16_kernel<<<dim3((unsigned)((n3 + 255) / 256)), 256, 0, stream>>>(wo, wo_bf, n3);
    cvt_wb1t_kernel<<<dim3((H_ * KLR_ * NOPE_ + 255) / 256), 256, 0, stream>>>(wkv_b, wb1t);
    cvt_wb2_kernel<<<dim3((H_ * VHD_ * KLR_ + 255) / 256), 256, 0, stream>>>(wkv_b, wb2);
  }

  // ---- q = x @ wq^T  (M=4096, N=3072, K=2048) ----
  gemm128_kernel<0><<<dim3((H_ * QKHD_ + 127) / 128, M / 128, 1), 256, 0, stream>>>(
      x, DIM_, 0, wq_bf, DIM_, 0, qf, H_ * QKHD_, 0, nullptr, 1.0f,
      M, H_ * QKHD_, DIM_);

  // ---- kv = x @ wkv_a^T  (N=576) ----
  gemm128_kernel<0><<<dim3((KCAT_ + 127) / 128, M / 128, 1), 256, 0, stream>>>(
      x, DIM_, 0, wkva_bf, DIM_, 0, kvraw, KCAT_, 0, nullptr, 1.0f,
      M, KCAT_, DIM_);

  // ---- RMSNorm + RoPE(k_pe) -> kcat, kv_c^T ----
  prep_kv_kernel<<<dim3(M), 256, 0, stream>>>(kvraw, kvw, fcos, fsin, kcat, kvcT);

  // ---- RoPE(q_pe)*SCALE -> qcat[...,512:576] ----
  {
    const long tot = (long)B_ * S_ * H_ * 32;
    prep_q_kernel<<<dim3((unsigned)((tot + 255) / 256)), 256, 0, stream>>>(qf, fcos, fsin, qcat);
  }

  // ---- q_proj (batched over heads): qcat[...,0:512] = SCALE * q_nope @ wb1^T ----
  gemm128_kernel<1><<<dim3(KLR_ / 128, M / 128, H_), 256, 0, stream>>>(
      qf, H_ * QKHD_, QKHD_, wb1t, NOPE_, (long)KLR_ * NOPE_,
      nullptr, 0, 0, qcat, SCALE_, M, KLR_, NOPE_);

  // ---- flash attention in latent space ----
  attn_kernel<<<dim3(S_ / 64, H_, B_), 256, 0, stream>>>(qcat, kcat, kvcT, oat);

  // ---- per-head output projection: o2[:, h*128+d] = o_h @ wb2_h^T ----
  gemm128_kernel<0><<<dim3(1, M / 128, H_), 256, 0, stream>>>(
      oat, H_ * KLR_, KLR_, wb2, KLR_, (long)VHD_ * KLR_,
      o2, H_ * VHD_, VHD_, nullptr, 1.0f, M, VHD_, KLR_);

  // ---- final: out = o2 @ wo^T  (M=4096, N=2048, K=2048) ----
  gemm128_kernel<0><<<dim3(DIM_ / 128, M / 128, 1), 256, 0, stream>>>(
      o2, H_ * VHD_, 0, wo_bf, H_ * VHD_, 0, out, DIM_, 0, nullptr, 1.0f,
      M, DIM_, H_ * VHD_);
}
